// CGNNExpert_47304769798447
// MI455X (gfx1250) — compile-verified, matrix operand order checked by reading, not run on hardware
//
#include <hip/hip_runtime.h>

// ======================= problem constants (fixed by reference) ==============
#define NN      50000
#define EE      800000
#define IN_DIM  64
#define HID     128
#define HEADS   16
#define DH      8
#define LAYERS  3
#define NSTORES 50
#define ET      (EE + NN)          // real edges + one self loop per node
#define EPSLN   1e-5f
#define NT      (HID / 16)         // 8 column tiles

typedef __bf16 bf16_t;
typedef __attribute__((ext_vector_type(16))) __bf16 v16bf;
typedef __attribute__((ext_vector_type(8)))  float  v8f;

// ======================= WMMA fragment loaders ==============================
// A fragment, 16x32 bf16 tile (MxK), row-major source, row stride ld elements.
// Lanes 0-15: M=lane, elems 0..7 -> K=k0..k0+7, elems 8..15 -> K=k0+16..k0+23.
// Lanes 16-31: same M, K pattern shifted by +8 (per CDNA5 ISA 7.12.2).
__device__ inline v16bf load_a_bf16(const bf16_t* p, int ld, int row0, int k0, int lane) {
  int m  = lane & 15;
  int kb = (lane & 16) ? 8 : 0;
  const bf16_t* r = p + (size_t)(row0 + m) * ld + (k0 + kb);
  v16bf a;
#pragma unroll
  for (int i = 0; i < 8; ++i) { a[i] = r[i]; a[8 + i] = r[16 + i]; }
  return a;
}

// Same, but the source matrix is fp32 (converted to bf16 on the fly).
__device__ inline v16bf load_a_f32(const float* p, int ld, int row0, int k0, int lane) {
  int m  = lane & 15;
  int kb = (lane & 16) ? 8 : 0;
  const float* r = p + (size_t)(row0 + m) * ld + (k0 + kb);
  v16bf a;
#pragma unroll
  for (int i = 0; i < 8; ++i) { a[i] = (bf16_t)r[i]; a[8 + i] = (bf16_t)r[16 + i]; }
  return a;
}

// B fragment from a pre-packed buffer laid out in exact fragment order:
// pk[((kt*NT + nt)*32 + lane)*16] -> one aligned 32B vector load per lane.
__device__ inline v16bf load_b_pk(const bf16_t* pk, int kt, int nt, int lane) {
  return *(const v16bf*)(pk + ((((size_t)kt * NT + nt) * 32 + lane) << 4));
}

__device__ inline v8f wmma_bf16(v16bf a, v16bf b, v8f c) {
  return __builtin_amdgcn_wmma_f32_16x16x32_bf16(false, a, false, b, (short)0, c, false, false);
}

__device__ inline float lrelu02(float v) { return v > 0.f ? v : 0.2f * v; }

__device__ inline float warp_sum(float v) {
#pragma unroll
  for (int off = 16; off; off >>= 1) v += __shfl_xor(v, off, 32);
  return v;
}

// monotone float<->uint encoding for atomicMax-based segment max
__device__ inline unsigned enc_ord(float f) {
  unsigned u = __float_as_uint(f);
  return (u & 0x80000000u) ? ~u : (u | 0x80000000u);
}
__device__ inline float dec_ord(unsigned u) {
  u = (u & 0x80000000u) ? (u ^ 0x80000000u) : ~u;
  return __uint_as_float(u);
}

// ======================= utility kernels ====================================
__global__ void k_zero(unsigned* p, int n) {
  int i = blockIdx.x * 256 + threadIdx.x;
  if (i < n) p[i] = 0u;
}

// Convert a KxHID fp32 weight to bf16 AND repack into WMMA B-fragment order:
// out[((kt*NT + nt)*32 + lane)*16 + j]. One thread per (kt, nt, lane).
__global__ void k_pack_b(const float* __restrict__ W, bf16_t* __restrict__ pk, int K) {
  int t = blockIdx.x * 256 + threadIdx.x;
  int total = (K / 32) * NT * 32;
  if (t >= total) return;
  int lane = t & 31;
  int nt   = (t >> 5) % NT;
  int kt   = t / (32 * NT);
  int n    = nt * 16 + (lane & 15);
  int kb   = (lane & 16) ? 8 : 0;
  int k0   = kt * 32;
  bf16_t* o = pk + (size_t)t * 16;
#pragma unroll
  for (int i = 0; i < 8; ++i) {
    o[i]     = (bf16_t)W[(size_t)(k0 + kb + i) * HID + n];
    o[8 + i] = (bf16_t)W[(size_t)(k0 + 16 + kb + i) * HID + n];
  }
}

// degree + incoming-edge-weight sum (for self-loop attr = mean)
__global__ void k_deg(const int* dst, const float* ew, float* deg, float* wsum, int e) {
  int i = blockIdx.x * 256 + threadIdx.x;
  if (i >= e) return;
  int d = dst[i];
  atomicAdd(deg + d, 1.0f);
  atomicAdd(wsum + d, ew[i]);
}
__global__ void k_loopattr(const float* deg, const float* wsum, float* la, int n) {
  int i = blockIdx.x * 256 + threadIdx.x;
  if (i < n) la[i] = wsum[i] / fmaxf(deg[i], 1.0f);
}

// ======================= input projection: WMMA GEMM + GELU + LayerNorm =====
// block = 256 threads = 8 waves; wave w computes the 16x16 tile at cols w*16;
// the block produces a full 16x128 slab, then LayerNorm(GELU(.)) in LDS.
__global__ __launch_bounds__(256) void k_inproj(const float* __restrict__ x,
                                                const bf16_t* __restrict__ Wpk,  // packed 64x128
                                                const float* __restrict__ bias,
                                                const float* __restrict__ g,
                                                const float* __restrict__ beta,
                                                float* __restrict__ h,
                                                bf16_t* __restrict__ hb) {
  __shared__ float slab[16][HID + 8];
  const int lane = threadIdx.x & 31, wid = threadIdx.x >> 5;
  const int row0 = blockIdx.x * 16, nt = wid;

  v8f c = {0.f, 0.f, 0.f, 0.f, 0.f, 0.f, 0.f, 0.f};
#pragma unroll
  for (int kt = 0; kt < IN_DIM / 32; ++kt) {
    v16bf a = load_a_f32(x, IN_DIM, row0, kt * 32, lane);
    v16bf b = load_b_pk(Wpk, kt, nt, lane);
    c = wmma_bf16(a, b, c);
  }
  const int mhi = (lane & 16) ? 8 : 0;
  const int nn  = nt * 16 + (lane & 15);
#pragma unroll
  for (int r = 0; r < 8; ++r) slab[r + mhi][nn] = c[r] + bias[nn];
  __syncthreads();

  // LayerNorm(GELU(.)): one wave per row (rows wid and wid+8), 4 cols/lane
#pragma unroll
  for (int rr = 0; rr < 2; ++rr) {
    int row = wid + rr * 8;
    float v[4], s1 = 0.f, s2 = 0.f;
#pragma unroll
    for (int i = 0; i < 4; ++i) {
      float t = slab[row][lane * 4 + i];
      t = 0.5f * t * (1.0f + erff(t * 0.70710678118f));   // exact GELU
      v[i] = t; s1 += t; s2 += t * t;
    }
    s1 = warp_sum(s1); s2 = warp_sum(s2);
    float mu = s1 * (1.0f / HID), var = s2 * (1.0f / HID) - mu * mu;
    float rs = rsqrtf(var + EPSLN);
    size_t base = (size_t)(row0 + row) * HID;
#pragma unroll
    for (int i = 0; i < 4; ++i) {
      int cc = lane * 4 + i;
      float o = (v[i] - mu) * rs * g[cc] + beta[cc];
      h[base + cc]  = o;
      hb[base + cc] = (bf16_t)o;
    }
  }
}

// ======================= xl / xr projections: dual WMMA GEMM ================
// xl = h@Wl + bl, xr = h@Wr + br; A fragment reused for both B operands.
__global__ __launch_bounds__(256) void k_xlxr(const bf16_t* __restrict__ hb,
                                              const bf16_t* __restrict__ Wlpk,
                                              const bf16_t* __restrict__ Wrpk,
                                              const float* __restrict__ bl,
                                              const float* __restrict__ br,
                                              float* __restrict__ xl,
                                              float* __restrict__ xr) {
  const int lane = threadIdx.x & 31, wid = threadIdx.x >> 5;
  const int row0 = blockIdx.x * 16, nt = wid;
  v8f cl = {0.f, 0.f, 0.f, 0.f, 0.f, 0.f, 0.f, 0.f};
  v8f cr = cl;
#pragma unroll
  for (int kt = 0; kt < HID / 32; ++kt) {
    v16bf a  = load_a_bf16(hb, HID, row0, kt * 32, lane);
    v16bf b1 = load_b_pk(Wlpk, kt, nt, lane);
    cl = wmma_bf16(a, b1, cl);
    v16bf b2 = load_b_pk(Wrpk, kt, nt, lane);
    cr = wmma_bf16(a, b2, cr);
  }
  const int mhi = (lane & 16) ? 8 : 0;
  const int nn  = nt * 16 + (lane & 15);
#pragma unroll
  for (int r = 0; r < 8; ++r) {
    size_t idx = (size_t)(row0 + r + mhi) * HID + nn;
    xl[idx] = cl[r] + bl[nn];
    xr[idx] = cr[r] + br[nn];
  }
}

// ======================= edge phase A: logits + segment max =================
// one thread per (edge, head); edges >= EE are the per-node self loops.
__global__ void k_edge_logit(const int* __restrict__ src, const int* __restrict__ dst,
                             const float* __restrict__ ew, const float* __restrict__ la,
                             const float* __restrict__ xl, const float* __restrict__ xr,
                             const float* __restrict__ We, const float* __restrict__ att,
                             float* __restrict__ logit, unsigned* __restrict__ mx) {
  int t = blockIdx.x * 256 + threadIdx.x;
  if (t >= ET * HEADS) return;
  int e = t >> 4, hh = t & 15;
  int s, d; float ea;
  if (e < EE) { s = src[e]; d = dst[e]; ea = ew[e]; }
  else        { s = d = e - EE; ea = la[s]; }
  const float4* pl = (const float4*)(xl + (size_t)s * HID + hh * DH);
  const float4* pr = (const float4*)(xr + (size_t)d * HID + hh * DH);
  const float4* pw = (const float4*)(We + hh * DH);
  const float4* pa = (const float4*)(att + hh * DH);
  float4 l0 = pl[0], l1 = pl[1], r0 = pr[0], r1 = pr[1];
  float4 w0 = pw[0], w1 = pw[1], a0 = pa[0], a1 = pa[1];
  float lg = lrelu02(l0.x + r0.x + ea * w0.x) * a0.x
           + lrelu02(l0.y + r0.y + ea * w0.y) * a0.y
           + lrelu02(l0.z + r0.z + ea * w0.z) * a0.z
           + lrelu02(l0.w + r0.w + ea * w0.w) * a0.w
           + lrelu02(l1.x + r1.x + ea * w1.x) * a1.x
           + lrelu02(l1.y + r1.y + ea * w1.y) * a1.y
           + lrelu02(l1.z + r1.z + ea * w1.z) * a1.z
           + lrelu02(l1.w + r1.w + ea * w1.w) * a1.w;
  logit[t] = lg;
  atomicMax(mx + (size_t)d * HEADS + hh, enc_ord(lg));
}

// ======================= edge phase B: exp + segment sum ====================
__global__ void k_edge_exp(const int* __restrict__ dst, float* __restrict__ logit,
                           const unsigned* __restrict__ mx, float* __restrict__ Z) {
  int t = blockIdx.x * 256 + threadIdx.x;
  if (t >= ET * HEADS) return;
  int e = t >> 4, hh = t & 15;
  int d = (e < EE) ? dst[e] : (e - EE);
  float p = expf(logit[t] - dec_ord(mx[(size_t)d * HEADS + hh]));
  logit[t] = p;                                     // reuse buffer for p
  atomicAdd(Z + (size_t)d * HEADS + hh, p);
}

// ======================= edge phase C: alpha * xl[src] scatter ==============
// one thread per (edge, channel); 128 channels = 2 edges per 256-thread block.
__global__ void k_edge_scatter(const int* __restrict__ src, const int* __restrict__ dst,
                               const float* __restrict__ p, const float* __restrict__ Z,
                               const float* __restrict__ xl, float* __restrict__ agg) {
  int t = blockIdx.x * 256 + threadIdx.x;
  if (t >= ET * HID) return;
  int e = t >> 7, cc = t & 127, hh = cc >> 3;
  int s, d;
  if (e < EE) { s = src[e]; d = dst[e]; } else { s = d = e - EE; }
  float alpha = p[(size_t)e * HEADS + hh] / (Z[(size_t)d * HEADS + hh] + 1e-16f);
  atomicAdd(agg + (size_t)d * HID + cc, alpha * xl[(size_t)s * HID + cc]);
}

// ======================= residual + LayerNorm (per GAT layer) ===============
__global__ __launch_bounds__(256) void k_ln_res(float* __restrict__ h, bf16_t* __restrict__ hb,
                                                const float* __restrict__ agg,
                                                const float* __restrict__ bias,
                                                const float* __restrict__ g,
                                                const float* __restrict__ beta, int n) {
  const int lane = threadIdx.x & 31, wid = threadIdx.x >> 5;
  const int row = blockIdx.x * 8 + wid;
  if (row >= n) return;
  float* hr = h + (size_t)row * HID;
  const float* ar = agg + (size_t)row * HID;
  float v[4], s1 = 0.f, s2 = 0.f;
#pragma unroll
  for (int i = 0; i < 4; ++i) {
    int cc = lane * 4 + i;
    float t = hr[cc] + ar[cc] + bias[cc];
    v[i] = t; s1 += t; s2 += t * t;
  }
  s1 = warp_sum(s1); s2 = warp_sum(s2);
  float mu = s1 * (1.0f / HID), var = s2 * (1.0f / HID) - mu * mu;
  float rs = rsqrtf(var + EPSLN);
  bf16_t* hbr = hb + (size_t)row * HID;
#pragma unroll
  for (int i = 0; i < 4; ++i) {
    int cc = lane * 4 + i;
    float o = (v[i] - mu) * rs * g[cc] + beta[cc];
    hr[cc] = o; hbr[cc] = (bf16_t)o;
  }
}

// ======================= per-store mean / gate / final ======================
__global__ void k_store_sums(const float* __restrict__ h, const int* __restrict__ sm,
                             float* __restrict__ sums, float* __restrict__ cnt) {
  int t = blockIdx.x * 256 + threadIdx.x;
  if (t >= NN * HID) return;
  int node = t >> 7, cc = t & 127;
  int st = sm[node];
  atomicAdd(sums + (size_t)st * HID + cc, h[(size_t)node * HID + cc]);
  if (cc == 0) atomicAdd(cnt + st, 1.0f);
}

__global__ void k_means(const float* __restrict__ sums, const float* __restrict__ cnt,
                        float* __restrict__ means) {
  int t = blockIdx.x * 256 + threadIdx.x;
  if (t >= NSTORES * HID) return;
  means[t] = sums[t] / fmaxf(cnt[t >> 7], 1.0f);
}

// gate = sigmoid(means @ ctx_W + ctx_b); gm = gate * means. 50x128x128 -> scalar
// (0.8 MFLOP; WMMA setup cost would exceed the math).
__global__ void k_gate(const float* __restrict__ means, const float* __restrict__ W,
                       const float* __restrict__ b, float* __restrict__ gm) {
  int t = blockIdx.x * 256 + threadIdx.x;
  if (t >= NSTORES * HID) return;
  int s = t >> 7, cc = t & 127;
  float acc = b[cc];
  const float* mr = means + (size_t)s * HID;
#pragma unroll 4
  for (int k = 0; k < HID; ++k) acc += mr[k] * W[(size_t)k * HID + cc];
  float gate = 1.0f / (1.0f + expf(-acc));
  gm[t] = gate * mr[cc];
}

__global__ __launch_bounds__(256) void k_final(const float* __restrict__ h,
                                               const float* __restrict__ gm,
                                               const int* __restrict__ sm,
                                               const float* __restrict__ g,
                                               const float* __restrict__ b,
                                               float* __restrict__ out, int n) {
  const int lane = threadIdx.x & 31, wid = threadIdx.x >> 5;
  const int row = blockIdx.x * 8 + wid;
  if (row >= n) return;
  const float* hr = h + (size_t)row * HID;
  const float* gr = gm + (size_t)sm[row] * HID;
  float v[4], s1 = 0.f, s2 = 0.f;
#pragma unroll
  for (int i = 0; i < 4; ++i) {
    int cc = lane * 4 + i;
    float t = hr[cc] + gr[cc];
    v[i] = t; s1 += t; s2 += t * t;
  }
  s1 = warp_sum(s1); s2 = warp_sum(s2);
  float mu = s1 * (1.0f / HID), var = s2 * (1.0f / HID) - mu * mu;
  float rs = rsqrtf(var + EPSLN);
  float* orow = out + (size_t)row * HID;
#pragma unroll
  for (int i = 0; i < 4; ++i) {
    int cc = lane * 4 + i;
    orow[cc] = (v[i] - mu) * rs * g[cc] + b[cc];
  }
}

// ======================= host launcher ======================================
extern "C" void kernel_launch(void* const* d_in, const int* in_sizes, int n_in,
                              void* d_out, int out_size, void* d_ws, size_t ws_size,
                              hipStream_t stream) {
  (void)in_sizes; (void)n_in; (void)out_size; (void)ws_size;
  const float* x      = (const float*)d_in[0];
  const int*   eidx   = (const int*)  d_in[1];
  const float* ew     = (const float*)d_in[2];
  const int*   smask  = (const int*)  d_in[3];
  const float* in_W   = (const float*)d_in[4];
  const float* in_b   = (const float*)d_in[5];
  const float* in_g   = (const float*)d_in[6];
  const float* in_be  = (const float*)d_in[7];
  const float* gWl    = (const float*)d_in[8];
  const float* gbl    = (const float*)d_in[9];
  const float* gWr    = (const float*)d_in[10];
  const float* gbr    = (const float*)d_in[11];
  const float* gWe    = (const float*)d_in[12];
  const float* gatt   = (const float*)d_in[13];
  const float* gbias  = (const float*)d_in[14];
  const float* blk_g  = (const float*)d_in[15];
  const float* blk_b  = (const float*)d_in[16];
  const float* ctx_W  = (const float*)d_in[17];
  const float* ctx_b  = (const float*)d_in[18];
  const float* fin_g  = (const float*)d_in[19];
  const float* fin_b  = (const float*)d_in[20];
  float* out = (float*)d_out;

  const int* src = eidx;
  const int* dst = eidx + EE;

  // ---- workspace carve-out (256B aligned) ----
  char* ws = (char*)d_ws;
  size_t off = 0;
  auto take = [&](size_t bytes) -> void* {
    void* p = ws + off;
    off += (bytes + 255) & ~(size_t)255;
    return p;
  };
  float*    h     = (float*)   take((size_t)NN * HID * 4);
  bf16_t*   hb    = (bf16_t*)  take((size_t)NN * HID * 2);
  float*    xl    = (float*)   take((size_t)NN * HID * 4);
  float*    xr    = (float*)   take((size_t)NN * HID * 4);
  float*    logit = (float*)   take((size_t)ET * HEADS * 4);  // reused as p
  unsigned* mx    = (unsigned*)take((size_t)NN * HEADS * 4);
  float*    Z     = (float*)   take((size_t)NN * HEADS * 4);
  float*    agg   = (float*)   take((size_t)NN * HID * 4);
  float*    deg   = (float*)   take((size_t)NN * 4);
  float*    wsum  = (float*)   take((size_t)NN * 4);
  float*    la    = (float*)   take((size_t)NN * 4);
  bf16_t*   inWpk = (bf16_t*)  take((size_t)IN_DIM * HID * 2);
  bf16_t*   Wlpk  = (bf16_t*)  take((size_t)LAYERS * HID * HID * 2);
  bf16_t*   Wrpk  = (bf16_t*)  take((size_t)LAYERS * HID * HID * 2);
  float*    sums  = (float*)   take((size_t)NSTORES * HID * 4);
  float*    cnt   = (float*)   take((size_t)NSTORES * 4);
  float*    means = (float*)   take((size_t)NSTORES * HID * 4);
  float*    gm    = (float*)   take((size_t)NSTORES * HID * 4);

  auto gr = [](long long n) { return dim3((unsigned)((n + 255) / 256)); };

  // ---- re-init accumulators (graph-replay safe) ----
  k_zero<<<gr(NN), 256, 0, stream>>>((unsigned*)deg, NN);
  k_zero<<<gr(NN), 256, 0, stream>>>((unsigned*)wsum, NN);
  k_zero<<<gr(NSTORES * HID), 256, 0, stream>>>((unsigned*)sums, NSTORES * HID);
  k_zero<<<gr(NSTORES), 256, 0, stream>>>((unsigned*)cnt, NSTORES);

  // ---- weight conversion + repack into WMMA B-fragment order ----
  k_pack_b<<<gr((IN_DIM / 32) * NT * 32), 256, 0, stream>>>(in_W, inWpk, IN_DIM);
  for (int l = 0; l < LAYERS; ++l) {
    k_pack_b<<<gr((HID / 32) * NT * 32), 256, 0, stream>>>(
        gWl + (size_t)l * HID * HID, Wlpk + (size_t)l * HID * HID, HID);
    k_pack_b<<<gr((HID / 32) * NT * 32), 256, 0, stream>>>(
        gWr + (size_t)l * HID * HID, Wrpk + (size_t)l * HID * HID, HID);
  }

  // ---- self-loop edge attr (mean of incoming weights) ----
  k_deg<<<gr(EE), 256, 0, stream>>>(dst, ew, deg, wsum, EE);
  k_loopattr<<<gr(NN), 256, 0, stream>>>(deg, wsum, la, NN);

  // ---- input projection + GELU + LN (WMMA) ----   N % 16 == 0
  k_inproj<<<dim3(NN / 16), 256, 0, stream>>>(x, inWpk, in_b, in_g, in_be, h, hb);

  // ---- 3 GATv2 layers ----
  for (int l = 0; l < LAYERS; ++l) {
    k_zero<<<gr((long long)NN * HEADS), 256, 0, stream>>>(mx, NN * HEADS);
    k_zero<<<gr((long long)NN * HEADS), 256, 0, stream>>>((unsigned*)Z, NN * HEADS);
    k_zero<<<gr((long long)NN * HID), 256, 0, stream>>>((unsigned*)agg, NN * HID);

    k_xlxr<<<dim3(NN / 16), 256, 0, stream>>>(hb,
        Wlpk + (size_t)l * HID * HID, Wrpk + (size_t)l * HID * HID,
        gbl + (size_t)l * HID, gbr + (size_t)l * HID, xl, xr);

    k_edge_logit<<<gr((long long)ET * HEADS), 256, 0, stream>>>(
        src, dst, ew, la, xl, xr, gWe + (size_t)l * HID, gatt + (size_t)l * HEADS * DH,
        logit, mx);
    k_edge_exp<<<gr((long long)ET * HEADS), 256, 0, stream>>>(dst, logit, mx, Z);
    k_edge_scatter<<<gr((long long)ET * HID), 256, 0, stream>>>(src, dst, logit, Z, xl, agg);

    k_ln_res<<<dim3((NN + 7) / 8), 256, 0, stream>>>(h, hb, agg,
        gbias + (size_t)l * HID, blk_g + (size_t)l * HID, blk_b + (size_t)l * HID, NN);
  }

  // ---- per-store gate + final LN ----
  k_store_sums<<<gr((long long)NN * HID), 256, 0, stream>>>(h, smask, sums, cnt);
  k_means<<<gr(NSTORES * HID), 256, 0, stream>>>(sums, cnt, means);
  k_gate<<<gr(NSTORES * HID), 256, 0, stream>>>(means, ctx_W, ctx_b, gm);
  k_final<<<dim3((NN + 7) / 8), 256, 0, stream>>>(h, gm, smask, fin_g, fin_b, out, NN);
}
